// MultiHeadCausalAttention_84902913507671
// MI455X (gfx1250) — matrix operand order repaired and checked by reading, not compiled, not run on hardware
//
#include <hip/hip_runtime.h>

// ---------------------------------------------------------------------------
// Multi-head causal attention for gfx1250 (MI455X), bf16 WMMA pipeline.
//   pass 1: fp32 -> bf16 conversion (x, Wq, Wk, Wv, Wo)
//   pass 2: Q/K/V projections, bf16 WMMA GEMM (softmax scale folded into Q)
//   pass 3: flash attention, 32-query tiles per wave; masked work only on the
//           diagonal chunk; row-sums via ones-WMMA; row-max via DPP butterfly
//   pass 4: ctx @ Wo -> fp32 output
// ---------------------------------------------------------------------------

typedef __attribute__((ext_vector_type(16))) __bf16 v16bf;
typedef __attribute__((ext_vector_type(8)))  __bf16 v8bf;
typedef __attribute__((ext_vector_type(4)))  __bf16 v4bf;
typedef __attribute__((ext_vector_type(2)))  __bf16 v2bf;
typedef __attribute__((ext_vector_type(8)))  float  v8f;
typedef __attribute__((ext_vector_type(4)))  float  v4f;

#define D_MODEL 1024
#define SEQ     2048
#define BATCH   2
#define NHEAD   16
#define HDIM    64
#define MTOT    (BATCH * SEQ)   // 4096 rows

static __device__ __forceinline__ v8f zero8() {
  v8f z = {0.f, 0.f, 0.f, 0.f, 0.f, 0.f, 0.f, 0.f};
  return z;
}

static __device__ __forceinline__ v8f wmma_bf16(v16bf a, v16bf b, v8f c) {
  // D = A(16x32 bf16) * B(32x16 bf16) + C(16x16 f32)
  return __builtin_amdgcn_wmma_f32_16x16x32_bf16(false, a, false, b,
                                                 (short)0, c, false, false);
}

// A-fragment (16x32 bf16): lane = g*16 + r16; row = r16.
// g=0: K = {0..15}; g=1: K = {16..31}
static __device__ __forceinline__ v16bf load_afrag(const __bf16* rowptr, int g) {
  v8bf lo = *(const v8bf*)(rowptr + g * 16);
  v8bf hi = *(const v8bf*)(rowptr + g * 16 + 8);
  v16bf o;
#pragma unroll
  for (int i = 0; i < 8; ++i) { o[i] = lo[i]; o[i + 8] = hi[i]; }
  return o;
}

// B-fragment (32x16 bf16): lane col n = r16, K = g*16 + e, e contiguous.
static __device__ __forceinline__ v16bf load_bfrag(const __bf16* p) {
  v8bf lo = *(const v8bf*)p;
  v8bf hi = *(const v8bf*)(p + 8);
  v16bf o;
#pragma unroll
  for (int i = 0; i < 8; ++i) { o[i] = lo[i]; o[i + 8] = hi[i]; }
  return o;
}

// ---------------------------------------------------------------------------
// 16-lane max reduction, pure VALU via DPP (no LDS, no dependency waits).
// Steps: quad xor1, quad xor2, row_half_mirror (combine quads), row_mirror
// (combine 8-groups). For a max-reduction any covering permutation works.
// ---------------------------------------------------------------------------
#if __has_builtin(__builtin_amdgcn_mov_dpp)
template <int N, int CTRL>
static __device__ __forceinline__ void redmax_step(float (&v)[N]) {
#pragma unroll
  for (int i = 0; i < N; ++i) {
    int t = __builtin_amdgcn_mov_dpp(__builtin_bit_cast(int, v[i]),
                                     CTRL, 0xF, 0xF, true);
    v[i] = fmaxf(v[i], __builtin_bit_cast(float, t));
  }
}
template <int N>
static __device__ __forceinline__ void redmax(float (&v)[N]) {
  redmax_step<N, 0xB1>(v);   // quad_perm [1,0,3,2]
  redmax_step<N, 0x4E>(v);   // quad_perm [2,3,0,1]
  redmax_step<N, 0x141>(v);  // row_half_mirror
  redmax_step<N, 0x140>(v);  // row_mirror
}
#else
template <int N>
static __device__ __forceinline__ void redmax(float (&v)[N]) {
#pragma unroll
  for (int m = 1; m < 16; m <<= 1) {
    float t[N];
#pragma unroll
    for (int i = 0; i < N; ++i) t[i] = __shfl_xor(v[i], m, 32);
#pragma unroll
    for (int i = 0; i < N; ++i) v[i] = fmaxf(v[i], t[i]);
  }
}
#endif

// ---------------------------------------------------------------------------
// fp32 -> bf16 conversion (vectorized x4)
// ---------------------------------------------------------------------------
__global__ void cvt_f32_bf16(const float* __restrict__ in,
                             __bf16* __restrict__ out, int n4) {
  int i = blockIdx.x * 256 + threadIdx.x;
  if (i < n4) {
    v4f v = *(const v4f*)(in + (size_t)i * 4);
    v4bf o;
#pragma unroll
    for (int j = 0; j < 4; ++j) o[j] = (__bf16)v[j];
    *(v4bf*)(out + (size_t)i * 4) = o;
  }
}

// ---------------------------------------------------------------------------
// bf16 GEMM: out[M,N] = scale * (A[M,K] * B[K,N] + bias), f32 WMMA accumulate.
// Block tile 64(M) x 128(N), BK=64. 8 waves (2x4), each wave 32x32 output.
// ---------------------------------------------------------------------------
template <int OUTF32, int HASBIAS>
__global__ __launch_bounds__(256) void gemm_bf16_wmma(
    const __bf16* __restrict__ A, const __bf16* __restrict__ B,
    const float* __restrict__ bias, void* __restrict__ outp,
    int M, int N, int K, float oscale) {
  __shared__ __align__(16) __bf16 As[64][72];    // +8 pad: conflict-free b128
  __shared__ __align__(16) __bf16 Bt[128][72];   // B transposed (N x K)

  const int tid  = threadIdx.x;
  const int lane = tid & 31, wid = tid >> 5;
  const int wm = wid >> 2, wn = wid & 3;
  const int g = lane >> 4, r16 = lane & 15;
  const int bM = blockIdx.y * 64, bN = blockIdx.x * 128;

  v8f acc[2][2];
#pragma unroll
  for (int i = 0; i < 2; ++i)
#pragma unroll
    for (int j = 0; j < 2; ++j) acc[i][j] = zero8();

  const int arow = tid >> 2, aseg = (tid & 3) * 16;  // 64 rows x 4 segs of 16
  const int brow = tid >> 3, bcol = (tid & 7) * 16;  // 2x32 k-rows x 8 n-segs

  for (int k0 = 0; k0 < K; k0 += 64) {
    // Stage A (64x64) straight, B (64x128) transposed into LDS.
    const __bf16* asrc  = A + (size_t)(bM + arow) * K + k0 + aseg;
    const __bf16* bsrc0 = B + (size_t)(k0 + brow) * N + bN + bcol;
    const __bf16* bsrc1 = B + (size_t)(k0 + 32 + brow) * N + bN + bcol;
    v8bf a0  = *(const v8bf*)asrc;
    v8bf a1  = *(const v8bf*)(asrc + 8);
    v8bf b00 = *(const v8bf*)bsrc0;
    v8bf b01 = *(const v8bf*)(bsrc0 + 8);
    v8bf b10 = *(const v8bf*)bsrc1;
    v8bf b11 = *(const v8bf*)(bsrc1 + 8);
    // Prefetch next K-tile while this one is consumed (global_prefetch_b8).
    __builtin_prefetch(asrc + 64, 0, 1);
    __builtin_prefetch(bsrc0 + (size_t)64 * N, 0, 1);

    *(v8bf*)&As[arow][aseg]     = a0;
    *(v8bf*)&As[arow][aseg + 8] = a1;
#pragma unroll
    for (int j = 0; j < 8; ++j) {
      Bt[bcol + j][brow]          = b00[j];
      Bt[bcol + 8 + j][brow]      = b01[j];
      Bt[bcol + j][32 + brow]     = b10[j];
      Bt[bcol + 8 + j][32 + brow] = b11[j];
    }
    __syncthreads();

#pragma unroll
    for (int c = 0; c < 2; ++c) {
      v16bf af[2], bfr[2];
#pragma unroll
      for (int mi = 0; mi < 2; ++mi)
        af[mi] = load_afrag(&As[wm * 32 + mi * 16 + r16][c * 32], g);
#pragma unroll
      for (int ni = 0; ni < 2; ++ni)
        bfr[ni] = load_bfrag(&Bt[wn * 32 + ni * 16 + r16][c * 32 + g * 16]);
#pragma unroll
      for (int mi = 0; mi < 2; ++mi)
#pragma unroll
        for (int ni = 0; ni < 2; ++ni)
          acc[mi][ni] = wmma_bf16(af[mi], bfr[ni], acc[mi][ni]);
    }
    __syncthreads();
  }

  // Epilogue. C/D layout: VGPR r -> rows r / r+8 per half, N = lane%16.
#pragma unroll
  for (int ni = 0; ni < 2; ++ni) {
    const int col = bN + wn * 32 + ni * 16 + r16;
    const float bv = HASBIAS ? bias[col] : 0.0f;
#pragma unroll
    for (int mi = 0; mi < 2; ++mi) {
#pragma unroll
      for (int r = 0; r < 8; ++r) {
        const int row = bM + wm * 32 + mi * 16 + r + g * 8;
        const float v = (acc[mi][ni][r] + bv) * oscale;
        if (OUTF32)
          ((float*)outp)[(size_t)row * N + col] = v;
        else
          ((__bf16*)outp)[(size_t)row * N + col] = (__bf16)v;
      }
    }
  }
}

// ---------------------------------------------------------------------------
// Flash attention chunk: 32 queries (2 M-tiles) x 32 keys.
// Score columns are remapped even/odd so each lane's two probabilities are
// adjacent keys -> packed b32 P-store. Row sums come from a ones-WMMA.
// MASK=false for fully-valid bulk chunks (no compares at all).
// ---------------------------------------------------------------------------
template <bool MASK>
static __device__ __forceinline__ void attn_chunk(
    int k0, int q0, int lane, int g, int r16,
    const __bf16* __restrict__ Kp, const __bf16* __restrict__ Vp,
    const v16bf (&qa)[2][2], const v16bf& ones,
    float (&mi)[2][8], float (&li)[2][8], v8f (&acc)[2][4],
    __bf16 (*pb)[40], __bf16 (*vt)[40]) {
  // Stage V chunk transposed: lane owns key row (k0+lane), scatter by d.
  const __bf16* vrow = Vp + (size_t)(k0 + lane) * D_MODEL;
#pragma unroll
  for (int d0 = 0; d0 < 64; d0 += 8) {
    v8bf vv = *(const v8bf*)(vrow + d0);
#pragma unroll
    for (int j = 0; j < 8; ++j) vt[d0 + j][lane] = vv[j];
  }
  __builtin_prefetch(vrow + (size_t)32 * D_MODEL, 0, 1);

  // K B-fragments: s0 columns = even keys (k0+2n), s1 columns = odd keys.
  const __bf16* krow0 = Kp + (size_t)(k0 + 2 * r16) * D_MODEL;
  const __bf16* krow1 = Kp + (size_t)(k0 + 2 * r16 + 1) * D_MODEL;
  const v16bf kb00 = load_bfrag(krow0 + g * 16);
  const v16bf kb01 = load_bfrag(krow0 + 32 + g * 16);
  const v16bf kb10 = load_bfrag(krow1 + g * 16);
  const v16bf kb11 = load_bfrag(krow1 + 32 + g * 16);
  __builtin_prefetch(krow0 + (size_t)32 * D_MODEL, 0, 1);

  // Scores for both query M-tiles (8 WMMAs), softmax scale pre-folded into Q.
  v8f s00 = zero8(), s01 = zero8(), s10 = zero8(), s11 = zero8();
  s00 = wmma_bf16(qa[0][0], kb00, s00);
  s00 = wmma_bf16(qa[0][1], kb01, s00);
  s01 = wmma_bf16(qa[0][0], kb10, s01);
  s01 = wmma_bf16(qa[0][1], kb11, s01);
  s10 = wmma_bf16(qa[1][0], kb00, s10);
  s10 = wmma_bf16(qa[1][1], kb01, s10);
  s11 = wmma_bf16(qa[1][0], kb10, s11);
  s11 = wmma_bf16(qa[1][1], kb11, s11);

  float a0[2][8], a1[2][8], rm[16];
#pragma unroll
  for (int r = 0; r < 8; ++r) {
    a0[0][r] = s00[r]; a1[0][r] = s01[r];
    a0[1][r] = s10[r]; a1[1][r] = s11[r];
  }
  if (MASK) {
#pragma unroll
    for (int mt = 0; mt < 2; ++mt)
#pragma unroll
      for (int r = 0; r < 8; ++r) {
        const int q = q0 + mt * 16 + r + g * 8;
        if (k0 + 2 * r16 > q)     a0[mt][r] = -1.0e30f;
        if (k0 + 2 * r16 + 1 > q) a1[mt][r] = -1.0e30f;
      }
  }
#pragma unroll
  for (int mt = 0; mt < 2; ++mt)
#pragma unroll
    for (int r = 0; r < 8; ++r)
      rm[mt * 8 + r] = fmaxf(a0[mt][r], a1[mt][r]);

  redmax<16>(rm);  // DPP butterfly: pure VALU, no LDS round-trips

#pragma unroll
  for (int mt = 0; mt < 2; ++mt) {
#pragma unroll
    for (int r = 0; r < 8; ++r) {
      const float nm   = fmaxf(mi[mt][r], rm[mt * 8 + r]);
      const float corr = __expf(mi[mt][r] - nm);
      mi[mt][r] = nm;
      li[mt][r] *= corr;
#pragma unroll
      for (int n = 0; n < 4; ++n) acc[mt][n][r] *= corr;
      v2bf pp;
      pp[0] = (__bf16)__expf(a0[mt][r] - nm);  // key k0 + 2*r16
      pp[1] = (__bf16)__expf(a1[mt][r] - nm);  // key k0 + 2*r16 + 1
      *(v2bf*)&pb[mt * 16 + r + g * 8][2 * r16] = pp;
    }
  }

  // ctx += P(32x32) * V(32x64); li += P * ones (row sums in C layout).
  v16bf vb[4];
#pragma unroll
  for (int n = 0; n < 4; ++n) vb[n] = load_bfrag(&vt[n * 16 + r16][g * 16]);
#pragma unroll
  for (int mt = 0; mt < 2; ++mt) {
    const v16bf pa = load_afrag(&pb[mt * 16 + r16][0], g);
    const v8f ls = wmma_bf16(pa, ones, zero8());
#pragma unroll
    for (int r = 0; r < 8; ++r) li[mt][r] += ls[r];
#pragma unroll
    for (int n = 0; n < 4; ++n) acc[mt][n] = wmma_bf16(pa, vb[n], acc[mt][n]);
  }
}

// One wave per 32-query tile; key chunks of 32; bulk chunks unmasked.
__global__ __launch_bounds__(128) void attn_causal_wmma(
    const __bf16* __restrict__ Q, const __bf16* __restrict__ K,
    const __bf16* __restrict__ V, __bf16* __restrict__ O) {
  __shared__ __align__(16) __bf16 Pbuf[4][32][40];  // probs, per-wave
  __shared__ __align__(16) __bf16 Vt[4][64][40];    // V chunk transposed

  const int wid = threadIdx.x >> 5, lane = threadIdx.x & 31;
  const int g = lane >> 4, r16 = lane & 15;
  const int w  = blockIdx.x * 4 + wid;   // 2048 waves total
  const int qt = w & 63, bh = w >> 6;
  const int b = bh >> 4, h = bh & 15;
  const int q0 = qt * 32;

  const size_t base = (size_t)b * SEQ * D_MODEL + (size_t)h * HDIM;
  const __bf16* Qp = Q + base;
  const __bf16* Kp = K + base;
  const __bf16* Vp = V + base;
  __bf16* Op = O + base;

  // Q fragments (32 x 64), resident for the whole tile (scale pre-applied).
  v16bf qa[2][2];
#pragma unroll
  for (int mt = 0; mt < 2; ++mt) {
    const __bf16* qrow = Qp + (size_t)(q0 + mt * 16 + r16) * D_MODEL;
    qa[mt][0] = load_afrag(qrow, g);
    qa[mt][1] = load_afrag(qrow + 32, g);
  }

  v16bf ones;
#pragma unroll
  for (int i = 0; i < 16; ++i) ones[i] = (__bf16)1.0f;

  float mi[2][8], li[2][8];
  v8f acc[2][4];
#pragma unroll
  for (int mt = 0; mt < 2; ++mt)
#pragma unroll
    for (int r = 0; r < 8; ++r) { mi[mt][r] = -3.0e38f; li[mt][r] = 0.f; }
#pragma unroll
  for (int mt = 0; mt < 2; ++mt)
#pragma unroll
    for (int n = 0; n < 4; ++n) acc[mt][n] = zero8();

  __bf16 (*pb)[40] = Pbuf[wid];
  __bf16 (*vt)[40] = Vt[wid];

  // Bulk chunks: every key <= every query in this tile -> no mask compares.
  for (int k0 = 0; k0 < q0; k0 += 32)
    attn_chunk<false>(k0, q0, lane, g, r16, Kp, Vp, qa, ones, mi, li, acc, pb, vt);
  // Diagonal chunk: keys q0 .. q0+31, causal mask applied.
  attn_chunk<true>(q0, q0, lane, g, r16, Kp, Vp, qa, ones, mi, li, acc, pb, vt);

  // Normalize (fast reciprocal) and write ctx (bf16).
#pragma unroll
  for (int mt = 0; mt < 2; ++mt)
#pragma unroll
    for (int r = 0; r < 8; ++r) {
      const float inv = __builtin_amdgcn_rcpf(li[mt][r]);
#pragma unroll
      for (int n = 0; n < 4; ++n) {
        const float o = acc[mt][n][r] * inv;
        Op[(size_t)(q0 + mt * 16 + r + g * 8) * D_MODEL + n * 16 + r16] =
            (__bf16)o;
      }
    }
}

// ---------------------------------------------------------------------------
// Launch
// ---------------------------------------------------------------------------
extern "C" void kernel_launch(void* const* d_in, const int* in_sizes, int n_in,
                              void* d_out, int out_size, void* d_ws, size_t ws_size,
                              hipStream_t stream) {
  (void)in_sizes; (void)n_in; (void)out_size; (void)ws_size;
  const float* x  = (const float*)d_in[0];
  const float* Wq = (const float*)d_in[1];
  const float* bq = (const float*)d_in[2];
  const float* Wk = (const float*)d_in[3];
  const float* bk = (const float*)d_in[4];
  const float* Wv = (const float*)d_in[5];
  const float* bv = (const float*)d_in[6];
  const float* Wo = (const float*)d_in[7];

  char* ws = (char*)d_ws;
  const size_t MB = 1024u * 1024u;
  __bf16* xb  = (__bf16*)(ws + 0 * MB);   // 8 MB  (4096 x 1024 bf16)
  __bf16* Qb  = (__bf16*)(ws + 8 * MB);   // 8 MB (pre-scaled by Dh^-1/2)
  __bf16* Kb  = (__bf16*)(ws + 16 * MB);  // 8 MB
  __bf16* Vb  = (__bf16*)(ws + 24 * MB);  // 8 MB
  __bf16* Cb  = (__bf16*)(ws + 32 * MB);  // 8 MB (ctx)
  __bf16* wqb = (__bf16*)(ws + 40 * MB);  // 2 MB each
  __bf16* wkb = (__bf16*)(ws + 42 * MB);
  __bf16* wvb = (__bf16*)(ws + 44 * MB);
  __bf16* wob = (__bf16*)(ws + 46 * MB);

  const int nx4 = MTOT * D_MODEL / 4;
  const int nw4 = D_MODEL * D_MODEL / 4;
  cvt_f32_bf16<<<(nx4 + 255) / 256, 256, 0, stream>>>(x,  xb,  nx4);
  cvt_f32_bf16<<<(nw4 + 255) / 256, 256, 0, stream>>>(Wq, wqb, nw4);
  cvt_f32_bf16<<<(nw4 + 255) / 256, 256, 0, stream>>>(Wk, wkb, nw4);
  cvt_f32_bf16<<<(nw4 + 255) / 256, 256, 0, stream>>>(Wv, wvb, nw4);
  cvt_f32_bf16<<<(nw4 + 255) / 256, 256, 0, stream>>>(Wo, wob, nw4);

  dim3 gg(D_MODEL / 128, MTOT / 64);
  const float qscale = 0.125f;  // HEAD_DIM^-0.5, folded into Q projection
  gemm_bf16_wmma<0, 1><<<gg, 256, 0, stream>>>(xb, wqb, bq, (void*)Qb, MTOT, D_MODEL, D_MODEL, qscale);
  gemm_bf16_wmma<0, 1><<<gg, 256, 0, stream>>>(xb, wkb, bk, (void*)Kb, MTOT, D_MODEL, D_MODEL, 1.0f);
  gemm_bf16_wmma<0, 1><<<gg, 256, 0, stream>>>(xb, wvb, bv, (void*)Vb, MTOT, D_MODEL, D_MODEL, 1.0f);

  attn_causal_wmma<<<(BATCH * NHEAD * (SEQ / 32)) / 4, 128, 0, stream>>>(Qb, Kb, Vb, Cb);

  gemm_bf16_wmma<1, 0><<<gg, 256, 0, stream>>>(Cb, wob, nullptr, d_out, MTOT, D_MODEL, D_MODEL, 1.0f);
}
